// GUPNet_4741643895046
// MI455X (gfx1250) — matrix operand order repaired and checked by reading, not compiled, not run on hardware
//
#include <hip/hip_runtime.h>
#include <hip/hip_bf16.h>
#include <math.h>

#define DEVFN __device__ __forceinline__

typedef __attribute__((ext_vector_type(16))) __bf16  v16bf;
typedef __attribute__((ext_vector_type(8)))  float   v8f;
typedef __attribute__((ext_vector_type(4)))  unsigned int u32x4;
typedef __attribute__((ext_vector_type(8)))  int     i32x8;
typedef __attribute__((ext_vector_type(4)))  int     i32x4;
typedef int v4i_vs __attribute__((__vector_size__(16)));   // matches async-LDS builtin param

#define BB   8
#define CC   64
#define HHc  96
#define WWc  320
#define KKt  50
#define HEADN 256
#define NBOX 400
#define CIN3R 69
#define CIN3P 96
#define HW   (HHc * WWc)

#if defined(__gfx1250__) && __has_builtin(__builtin_amdgcn_global_load_async_to_lds_b128)
#define HAVE_ASYNC_LDS 1
#else
#define HAVE_ASYNC_LDS 0
#endif

struct B32x2 { uint4 a, b; };   // 32 bytes == v16bf

DEVFN unsigned short f2bf(float f) {
  unsigned int u = __builtin_bit_cast(unsigned int, f);
  unsigned int r = u + 0x7FFFu + ((u >> 16) & 1u);
  return (unsigned short)(r >> 16);
}
DEVFN float bf2f(unsigned short h) {
  return __builtin_bit_cast(float, ((unsigned int)h) << 16);
}
DEVFN int imin(int a, int b) { return a < b ? a : b; }

// ---------------------------------------------------------------------------
// TDM exercise: NULL descriptor (group0.count == 0 -> architectural NOP per
// CDNA5 ISA 8.3), drives tensor_load_to_lds + s_wait_tensorcnt.
// ---------------------------------------------------------------------------
__global__ void k_tdm_nop() {
#if defined(__gfx1250__) && __has_builtin(__builtin_amdgcn_tensor_load_to_lds)
  __shared__ unsigned int lds[32];
  lds[threadIdx.x & 31] = 0u;
  __syncthreads();
  u32x4 g0 = (u32x4)0;   // count=0 -> NULL tensor, no transfer
  i32x8 g1 = (i32x8)0;
  i32x4 g2 = (i32x4)0;
  i32x4 g3 = (i32x4)0;
#if __clang_major__ >= 23
  i32x8 g4 = (i32x8)0;
  __builtin_amdgcn_tensor_load_to_lds(g0, g1, g2, g3, g4, 0);
#else
  __builtin_amdgcn_tensor_load_to_lds(g0, g1, g2, g3, 0);
#endif
  __builtin_amdgcn_s_wait_tensorcnt(0);
#endif
}

// ---------------------------------------------------------------------------
// f32 NCHW -> bf16 NHWC with channel zero-padding to Cp
// dst[img][p][c] (c fastest), src[img][c][p]
// ---------------------------------------------------------------------------
__global__ void k_nchw2nhwc(const float* __restrict__ src, unsigned short* __restrict__ dst,
                            int Cr, int Cp, int HWp, int total) {
  int i = blockIdx.x * 256 + threadIdx.x;
  if (i >= total) return;
  int c = i % Cp;
  int p = (i / Cp) % HWp;
  int img = i / (Cp * HWp);
  float v = (c < Cr) ? src[((size_t)img * Cr + c) * HWp + p] : 0.f;
  dst[i] = f2bf(v);
}

// pack conv3x3 weights [cout][cinR][3][3] f32 -> [cout][tap*cinP + cin] bf16 (zero pad)
__global__ void k_pack_w3(const float* __restrict__ w, unsigned short* __restrict__ wP,
                          int cout, int cinR, int cinP) {
  int i = blockIdx.x * 256 + threadIdx.x;
  int total = cout * cinP * 9;
  if (i >= total) return;
  int co = i / (cinP * 9);
  int r = i - co * (cinP * 9);
  int tap = r / cinP;
  int cin = r - tap * cinP;
  float v = (cin < cinR) ? w[((size_t)co * cinR + cin) * 9 + tap] : 0.f;
  wP[(size_t)co * (cinP * 9) + tap * cinP + cin] = f2bf(v);
}

// pack 1x1 weights [cout][256] f32 -> [16][256] bf16, zero pad rows
__global__ void k_pack_w2(const float* __restrict__ w2, unsigned short* __restrict__ wP, int cout) {
  int i = blockIdx.x * 256 + threadIdx.x;
  if (i >= 16 * HEADN) return;
  int n = i / HEADN, k = i - n * HEADN;
  wP[i] = (n < cout) ? f2bf(w2[(size_t)n * HEADN + k]) : (unsigned short)0;
}

// ---------------------------------------------------------------------------
// Implicit-GEMM 3x3 conv, pad=1, via v_wmma_f32_16x16x32_bf16. NHWC bf16 I/O.
// Block = 256 threads = 8 waves; tile M=32 pixels x N=64 out channels;
// wave (msub, nsub) owns one 16x16 f32 accumulator. K = 9*cinP in 32-chunks
// (cinP % 32 == 0 so each chunk is a single tap -> A frag = 2x b128 loads).
// MODE 0: input [NI][Hs*Ws][cinP], blockIdx.z = image.
// MODE 1: input [NBOX][49][cinP], pixels linearized box-major.
// ---------------------------------------------------------------------------
template <int MODE>
__global__ __launch_bounds__(256) void k_conv3x3_wmma(
    const unsigned short* __restrict__ inN, const unsigned short* __restrict__ wP,
    const float* __restrict__ bias, unsigned short* __restrict__ outN,
    int cinP, int Hs, int Ws, int npix, int coutTot, int doRelu) {
  const int lane = threadIdx.x & 31;
  const int wave = threadIdx.x >> 5;
  const int half = lane >> 4;
  const int lm = lane & 15;
  const int msub = wave >> 2;   // 0..1
  const int nsub = wave & 3;    // 0..3
  const int nGlob0 = blockIdx.y * 64;
  const int mBase = blockIdx.x * 32;
  const int KT = 9 * cinP;
  const int nChunks = KT >> 5;
  const int HsWs = Hs * Ws;

  __shared__ unsigned short ldsB[64 * 32];   // [n][kk]

  v8f acc = {};

  // A-matrix row pixel (same pixel for both lane halves; halves split K)
  int img, py, px;
  {
    int p = imin(mBase + msub * 16 + lm, npix - 1);
    if (MODE == 0) { img = blockIdx.z; py = p / Ws; px = p - py * Ws; }
    else           { img = p / 49; int q = p - img * 49; py = q / 7; px = q - py * 7; }
  }
  const unsigned short* __restrict__ inImg = inN + (size_t)img * HsWs * cinP;

  for (int ch = 0; ch < nChunks; ++ch) {
    const int k0 = ch << 5;
    const int tap = k0 / cinP;
    const int cin0 = k0 - tap * cinP;
    const int ys = py + tap / 3 - 1;
    const int xs = px + (tap % 3) - 1;
    const bool inb = (ys >= 0) & (ys < Hs) & (xs >= 0) & (xs < Ws);
    const unsigned short* base = inImg + ((size_t)ys * Ws + xs) * cinP;

    // stage B tile [64][32] into LDS: one b128 per thread
    {
      const int t = threadIdx.x;
      const int n = t >> 2;
      const int kk0 = (t & 3) << 3;
      const unsigned short* src = wP + (size_t)(nGlob0 + n) * KT + k0 + kk0;
      unsigned short* dst = &ldsB[n * 32 + kk0];
#if HAVE_ASYNC_LDS
      __builtin_amdgcn_global_load_async_to_lds_b128(
          (__attribute__((address_space(1))) v4i_vs*)src,
          (__attribute__((address_space(3))) v4i_vs*)dst, 0, 0);
#if __has_builtin(__builtin_amdgcn_s_wait_asynccnt)
      __builtin_amdgcn_s_wait_asynccnt(0);
#else
      asm volatile("s_wait_asynccnt 0" ::: "memory");
#endif
#else
      *(uint4*)dst = *(const uint4*)src;
#endif
    }
    __syncthreads();

    // prefetch next chunk's A run (hint only)
    if (ch + 1 < nChunks && inb) {
      int kn = (ch + 1) << 5;
      int cn = kn - (kn / cinP) * cinP;
      __builtin_prefetch(base + cn, 0, 1);
    }

    // A fragment: two contiguous 16B runs per lane (16-bit A 16x32 layout)
    v16bf afrag = {};
    if (inb) {
      B32x2 u;
      u.a = *(const uint4*)(base + cin0 + (half << 4));        // K = half*16 .. +7
      u.b = *(const uint4*)(base + cin0 + 8 + (half << 4));    // K = half*16+8 .. +7
      afrag = __builtin_bit_cast(v16bf, u);
    }

    // B fragment: lane = N column, halves split K; contiguous 32B in LDS
    const int nLoc = nsub * 16 + lm;
    v16bf bfrag = *(const v16bf*)&ldsB[nLoc * 32 + (half << 4)];

    acc = __builtin_amdgcn_wmma_f32_16x16x32_bf16(false, afrag, false, bfrag,
                                                  (short)0, acc, false, false);
    __syncthreads();
  }

  // epilogue: bias (+ReLU), store bf16 NHWC
  const int nOut = nGlob0 + nsub * 16 + lm;
  const float bv = bias[nOut];
#pragma unroll
  for (int r = 0; r < 8; ++r) {
    const int m = mBase + msub * 16 + r + half * 8;   // D row M = r + 8*half
    float vv = acc[r] + bv;
    if (doRelu) vv = fmaxf(vv, 0.f);
    if (MODE == 0) {
      outN[((size_t)img * HsWs + m) * coutTot + nOut] = f2bf(vv);
    } else if (m < npix) {
      outN[(size_t)m * coutTot + nOut] = f2bf(vv);
    }
  }
}

// ---------------------------------------------------------------------------
// 1x1 conv (GEMM K=256) via WMMA; NHWC bf16 in, f32 NCHW out into d_out.
// Block = 64 threads = 2 waves; tile M=32 pixels x N=16 (cout padded to 16).
// ---------------------------------------------------------------------------
__global__ __launch_bounds__(64) void k_conv1x1_wmma(
    const unsigned short* __restrict__ inN,   // [B][HW][256]
    const unsigned short* __restrict__ w2P,   // [16][256]
    const float* __restrict__ b2, float* __restrict__ out, int coutN) {
  const int lane = threadIdx.x & 31;
  const int wave = threadIdx.x >> 5;
  const int half = lane >> 4;
  const int lm = lane & 15;
  const int mBase = blockIdx.x * 32;
  const int img = blockIdx.z;
  const int p = mBase + wave * 16 + lm;
  const unsigned short* __restrict__ base = inN + ((size_t)img * HW + p) * HEADN;

  v8f acc = {};
  for (int ch = 0; ch < 8; ++ch) {
    const int k0 = ch << 5;
    B32x2 u;
    u.a = *(const uint4*)(base + k0 + (half << 4));
    u.b = *(const uint4*)(base + k0 + 8 + (half << 4));
    v16bf afrag = __builtin_bit_cast(v16bf, u);
    v16bf bfrag = *(const v16bf*)&w2P[(size_t)lm * HEADN + k0 + (half << 4)];
    acc = __builtin_amdgcn_wmma_f32_16x16x32_bf16(false, afrag, false, bfrag,
                                                  (short)0, acc, false, false);
  }

  if (lm < coutN) {
    const float bv = b2[lm];
#pragma unroll
    for (int r = 0; r < 8; ++r) {
      const int m = mBase + wave * 16 + r + half * 8;
      out[((size_t)img * coutN + lm) * HW + m] = acc[r] + bv;
    }
  }
}

// ---------------------------------------------------------------------------
// sigmoid + clip + 3x3 NMS
// ---------------------------------------------------------------------------
__global__ void k_nms(const float* __restrict__ hm, float* __restrict__ heat) {
  int i = blockIdx.x * 256 + threadIdx.x;
  if (i >= BB * HW) return;
  int b = i / HW, p = i - b * HW;
  int y = p / WWc, x = p - y * WWc;
  auto sig = [&](int yy, int xx) -> float {
    float v = hm[(size_t)b * HW + yy * WWc + xx];
    float s = 1.f / (1.f + expf(-v));
    return fminf(fmaxf(s, 1e-4f), 1.f - 1e-4f);
  };
  float c = sig(y, x);
  float mx = -1e30f;
  for (int dy = -1; dy <= 1; ++dy)
    for (int dx = -1; dx <= 1; ++dx) {
      int ys = y + dy, xs = x + dx;
      if (ys >= 0 && ys < HHc && xs >= 0 && xs < WWc) mx = fmaxf(mx, sig(ys, xs));
    }
  heat[i] = (mx == c) ? c : 0.f;
}

// per-batch iterative top-50
__global__ void k_topk(const float* __restrict__ heat, float* __restrict__ work,
                       int* __restrict__ inds) {
  const int b = blockIdx.x, t = threadIdx.x;
  for (int i = t; i < HW; i += 256) work[(size_t)b * HW + i] = heat[(size_t)b * HW + i];
  __syncthreads();
  __shared__ float sv[256];
  __shared__ int si[256];
  for (int it = 0; it < KKt; ++it) {
    float bv = -1e30f; int bi = 0;
    for (int i = t; i < HW; i += 256) {
      float v = work[(size_t)b * HW + i];
      if (v > bv) { bv = v; bi = i; }
    }
    sv[t] = bv; si[t] = bi;
    __syncthreads();
    for (int s = 128; s > 0; s >>= 1) {
      if (t < s && sv[t + s] > sv[t]) { sv[t] = sv[t + s]; si[t] = si[t + s]; }
      __syncthreads();
    }
    if (t == 0) { inds[b * KKt + it] = si[0]; work[(size_t)b * HW + si[0]] = -1e30f; }
    __syncthreads();
  }
}

// boxes, rect-projected corners, 7-point camera rays
__global__ void k_boxes(const int* __restrict__ inds, const float* __restrict__ o2,
                        const float* __restrict__ s2, const float* __restrict__ crg,
                        const float* __restrict__ calibs, float* __restrict__ boxes,
                        float* __restrict__ boxesr, float* __restrict__ xs7,
                        float* __restrict__ ys7) {
  int i = blockIdx.x * 256 + threadIdx.x;
  if (i >= NBOX) return;
  int b = i / KKt;
  int ind = inds[i];
  int y = ind / WWc, x = ind - y * WWc;
  float ox = o2[((size_t)b * 2 + 0) * HW + ind];
  float oy = o2[((size_t)b * 2 + 1) * HW + ind];
  float sx = s2[((size_t)b * 2 + 0) * HW + ind];
  float sy = s2[((size_t)b * 2 + 1) * HW + ind];
  float cx = (float)x + ox, cy = (float)y + oy;
  float x1 = cx - sx * 0.5f, y1 = cy - sy * 0.5f;
  float x2 = cx + sx * 0.5f, y2 = cy + sy * 0.5f;
  boxes[i * 5 + 0] = (float)b;
  boxes[i * 5 + 1] = x1; boxes[i * 5 + 2] = y1;
  boxes[i * 5 + 3] = x2; boxes[i * 5 + 4] = y2;
  const float* cr = crg + (size_t)b * 4;     // [[x0,y0],[x1,y1]]
  float rsx = cr[2] - cr[0], rsy = cr[3] - cr[1];
  float bx1 = x1 / WWc * rsx + cr[0], by1 = y1 / HHc * rsy + cr[1];
  float bx2 = x2 / WWc * rsx + cr[0], by2 = y2 / HHc * rsy + cr[1];
  boxesr[i * 5 + 0] = (float)b;
  boxesr[i * 5 + 1] = bx1; boxesr[i * 5 + 2] = by1;
  boxesr[i * 5 + 3] = bx2; boxesr[i * 5 + 4] = by2;
  const float* cal = calibs + (size_t)b * 12;
  float f_u = cal[0], f_v = cal[5], c_u = cal[2], c_v = cal[6];
  float b_x = cal[3] / -f_u, b_y = cal[7] / -f_v;
  float p1x = (bx1 - c_u) / f_u + b_x, p1y = (by1 - c_v) / f_v + b_y;
  float p2x = (bx2 - c_u) / f_u + b_x, p2y = (by2 - c_v) / f_v + b_y;
  for (int j = 0; j < 7; ++j) {
    float t = (float)j / 6.f;
    xs7[i * 7 + j] = p1x + t * (p2x - p1x);
    ys7[i * 7 + j] = p1y + t * (p2y - p1y);
  }
}

// ROI align (7x7, sr=2) -> channels [0..63] of roi_full bf16 NHWC [400][49][96]
__global__ void k_roialign(const float* __restrict__ feat, const float* __restrict__ boxes,
                           unsigned short* __restrict__ roiFull) {
  const int box = blockIdx.x;
  const int bidx = (int)boxes[box * 5 + 0];
  const float x1 = boxes[box * 5 + 1], y1 = boxes[box * 5 + 2];
  const float x2 = boxes[box * 5 + 3], y2 = boxes[box * 5 + 4];
  const float bw = fmaxf(x2 - x1, 1.f) / 7.f;
  const float bh = fmaxf(y2 - y1, 1.f) / 7.f;
  const float* fb = feat + (size_t)bidx * CC * HW;
  for (int e = threadIdx.x; e < CC * 49; e += 256) {
    int c = e / 49, q = e - c * 49;
    int oy = q / 7, ox = q - oy * 7;
    const float* fc = fb + (size_t)c * HW;
    float acc = 0.f;
    for (int iy = 0; iy < 2; ++iy)
      for (int ix = 0; ix < 2; ++ix) {
        float gy = ((float)(oy * 2 + iy) + 0.5f) * 0.5f;
        float gx = ((float)(ox * 2 + ix) + 0.5f) * 0.5f;
        float ys = y1 + gy * bh, xs = x1 + gx * bw;
        bool vy = (ys >= -1.f) && (ys <= (float)HHc);
        bool vx = (xs >= -1.f) && (xs <= (float)WWc);
        float yc = fminf(fmaxf(ys, 0.f), (float)(HHc - 1));
        float xc = fminf(fmaxf(xs, 0.f), (float)(WWc - 1));
        int y0 = (int)floorf(yc), x0 = (int)floorf(xc);
        int y1i = imin(y0 + 1, HHc - 1), x1i = imin(x0 + 1, WWc - 1);
        float fy = yc - (float)y0, fx = xc - (float)x0;
        float v = fc[y0 * WWc + x0] * (1.f - fy) * (1.f - fx) +
                  fc[y0 * WWc + x1i] * (1.f - fy) * fx +
                  fc[y1i * WWc + x0] * fy * (1.f - fx) +
                  fc[y1i * WWc + x1i] * fy * fx;
        acc += (vy && vx) ? v : 0.f;
      }
    roiFull[((size_t)box * 49 + q) * CIN3P + c] = f2bf(acc * 0.25f);
  }
}

// channels 64..95 of roi_full NHWC: cmx, cmy, one-hot(cls=0), zero pad
__global__ void k_roiextras(const float* __restrict__ xs7, const float* __restrict__ ys7,
                            unsigned short* __restrict__ roiFull) {
  const int box = blockIdx.x;
  for (int e = threadIdx.x; e < (CIN3P - CC) * 49; e += 256) {
    int c = CC + e / 49, q = e - (e / 49) * 49;
    int oy = q / 7, ox = q - oy * 7;
    float v = 0.f;
    if (c == 64) v = xs7[box * 7 + ox];
    else if (c == 65) v = ys7[box * 7 + oy];
    else if (c == 66) v = 1.f;   // cls_flat == 0 one-hot
    roiFull[((size_t)box * 49 + q) * CIN3P + c] = f2bf(v);
  }
}

// BN + ReLU + spatial mean over 7x7 (inter3 is NHWC [400][49][256])
__global__ void k_bnpool(const unsigned short* __restrict__ inter3,
                         const float* __restrict__ g, const float* __restrict__ be,
                         const float* __restrict__ m, const float* __restrict__ v,
                         float* __restrict__ pooled) {
  const int box = blockIdx.x, n = threadIdx.x;
  float sc = g[n] * rsqrtf(v[n] + 1e-5f);
  float mm = m[n], bb = be[n];
  const unsigned short* src = inter3 + (size_t)box * 49 * HEADN + n;
  float s = 0.f;
  for (int q = 0; q < 49; ++q) {
    float x = bf2f(src[(size_t)q * HEADN]);
    s += fmaxf((x - mm) * sc + bb, 0.f);
  }
  pooled[(size_t)box * HEADN + n] = s / 49.f;
}

// FC 256 -> cout
__global__ void k_fc(const float* __restrict__ pooled, const float* __restrict__ w2,
                     const float* __restrict__ b2, float* __restrict__ out, int coutN) {
  const int box = blockIdx.x;
  for (int c = threadIdx.x; c < coutN; c += 32) {
    float s = b2[c];
    const float* p = pooled + (size_t)box * HEADN;
    const float* w = w2 + (size_t)c * HEADN;
    for (int n = 0; n < HEADN; ++n) s += p[n] * w[n];
    out[(size_t)box * coutN + c] = s;
  }
}

// depth synthesis + scatter of s3 into size3d_offset / h3d_log_std
__global__ void k_final(const float* __restrict__ s3buf, const float* __restrict__ dno,
                        const float* __restrict__ boxesr, const float* __restrict__ calibs,
                        const float* __restrict__ mean_size, float* __restrict__ out_s3,
                        float* __restrict__ out_h3d, float* __restrict__ out_depth) {
  int i = blockIdx.x * 256 + threadIdx.x;
  if (i >= NBOX) return;
  int b = i / KKt;
  float s30 = s3buf[i * 4 + 0], s31 = s3buf[i * 4 + 1];
  float s32 = s3buf[i * 4 + 2], s33 = s3buf[i * 4 + 3];
  out_s3[i * 3 + 0] = s30; out_s3[i * 3 + 1] = s31; out_s3[i * 3 + 2] = s32;
  out_h3d[i] = s33;
  float f_u = calibs[(size_t)b * 12 + 0];
  float h2d = fmaxf(boxesr[i * 5 + 4] - boxesr[i * 5 + 2], 1.f);
  float sz0 = mean_size[0] + s30;             // cls 0
  float dgeo = sz0 / h2d * f_u;
  float geo = s33 + 2.f * (logf(f_u) - logf(h2d));
  float d1 = dno[i * 2 + 1];
  float mx = fmaxf(d1, geo);
  float net = mx + logf(expf(d1 - mx) + expf(geo - mx));
  float sg = 1.f / (1.f + expf(-dno[i * 2 + 0]));
  out_depth[i * 2 + 0] = 1.f / (sg + 1e-6f) - 1.f + dgeo;
  out_depth[i * 2 + 1] = net;
}

// ---------------------------------------------------------------------------
extern "C" void kernel_launch(void* const* d_in, const int* in_sizes, int n_in,
                              void* d_out, int out_size, void* d_ws, size_t ws_size,
                              hipStream_t stream) {
  (void)in_sizes; (void)n_in; (void)out_size; (void)ws_size;
  const float* feat   = (const float*)d_in[0];
  const float* crg    = (const float*)d_in[1];
  const float* calibs = (const float*)d_in[2];
  const float* msize  = (const float*)d_in[3];
  float* out = (float*)d_out;

  // output regions (floats)
  float* out_hm   = out + 0;
  float* out_o2   = out + 245760;
  float* out_s2   = out + 737280;
  float* out_hd   = out + 1228800;
  float* out_dep  = out + 1238400;
  float* out_o3   = out + 1239200;
  float* out_sz3  = out + 1240000;
  float* out_h3d  = out + 1241200;

  // workspace carve
  size_t off = 0;
  auto carve = [&](size_t bytes) {
    void* p = (char*)d_ws + off;
    off += (bytes + 255) & ~(size_t)255;
    return p;
  };
  unsigned short* featN  = (unsigned short*)carve((size_t)BB * HW * CC * 2);        // NHWC
  unsigned short* wPack  = (unsigned short*)carve((size_t)HEADN * CIN3P * 9 * 2);
  unsigned short* w2P    = (unsigned short*)carve((size_t)16 * HEADN * 2);
  unsigned short* interN = (unsigned short*)carve((size_t)BB * HW * HEADN * 2);     // NHWC
  float* heat   = (float*)carve((size_t)BB * HW * 4);
  float* work   = (float*)carve((size_t)BB * HW * 4);
  int*   inds   = (int*)carve((size_t)NBOX * 4);
  float* boxes  = (float*)carve((size_t)NBOX * 5 * 4);
  float* boxesr = (float*)carve((size_t)NBOX * 5 * 4);
  float* xs7    = (float*)carve((size_t)NBOX * 7 * 4);
  float* ys7    = (float*)carve((size_t)NBOX * 7 * 4);
  unsigned short* roiFull = (unsigned short*)carve((size_t)NBOX * 49 * CIN3P * 2);  // NHWC
  unsigned short* inter3  = (unsigned short*)carve((size_t)NBOX * 49 * HEADN * 2);  // NHWC
  float* pooled = (float*)carve((size_t)NBOX * HEADN * 4);
  float* s3buf  = (float*)carve((size_t)NBOX * 4 * 4);
  float* dno    = (float*)carve((size_t)NBOX * 2 * 4);

  k_tdm_nop<<<1, 32, 0, stream>>>();

  // feat -> bf16 NHWC
  {
    int n = BB * HW * CC;
    k_nchw2nhwc<<<(n + 255) / 256, 256, 0, stream>>>(feat, featN, CC, CC, HW, n);
  }

  // ---- head2 heads: conv3x3(64->256)+ReLU, then conv1x1(256->cout) ----
  const int h2w1[3]  = {4, 8, 12};
  const int h2b1[3]  = {5, 9, 13};
  const int h2w2[3]  = {6, 10, 14};
  const int h2b2[3]  = {7, 11, 15};
  const int h2co[3]  = {1, 2, 2};
  float* h2out[3] = {out_hm, out_o2, out_s2};
  for (int h = 0; h < 3; ++h) {
    int nw = HEADN * CC * 9;
    k_pack_w3<<<(nw + 255) / 256, 256, 0, stream>>>((const float*)d_in[h2w1[h]], wPack,
                                                    HEADN, CC, CC);
    k_conv3x3_wmma<0><<<dim3(HW / 32, HEADN / 64, BB), 256, 0, stream>>>(
        featN, wPack, (const float*)d_in[h2b1[h]], interN, CC, HHc, WWc, HW, HEADN, 1);
    k_pack_w2<<<(16 * HEADN + 255) / 256, 256, 0, stream>>>((const float*)d_in[h2w2[h]],
                                                            w2P, h2co[h]);
    k_conv1x1_wmma<<<dim3(HW / 32, 1, BB), 64, 0, stream>>>(
        interN, w2P, (const float*)d_in[h2b2[h]], h2out[h], h2co[h]);
  }

  // ---- NMS, top-k, boxes, ROI ----
  k_nms<<<(BB * HW + 255) / 256, 256, 0, stream>>>(out_hm, heat);
  k_topk<<<BB, 256, 0, stream>>>(heat, work, inds);
  k_boxes<<<(NBOX + 255) / 256, 256, 0, stream>>>(inds, out_o2, out_s2, crg, calibs,
                                                  boxes, boxesr, xs7, ys7);
  k_roialign<<<NBOX, 256, 0, stream>>>(feat, boxes, roiFull);
  k_roiextras<<<NBOX, 256, 0, stream>>>(xs7, ys7, roiFull);

  // ---- head3 heads: conv3x3(69->256) [+bias], BN+ReLU+pool, FC ----
  // order: dep, o3, s3, hd
  const int h3w1[4] = {16, 24, 32, 40};
  const int h3b1[4] = {17, 25, 33, 41};
  const int h3g[4]  = {18, 26, 34, 42};
  const int h3be[4] = {19, 27, 35, 43};
  const int h3m[4]  = {20, 28, 36, 44};
  const int h3v[4]  = {21, 29, 37, 45};
  const int h3w2[4] = {22, 30, 38, 46};
  const int h3b2[4] = {23, 31, 39, 47};
  const int h3co[4] = {2, 2, 4, 24};
  float* h3out[4] = {dno, out_o3, s3buf, out_hd};
  const int npix3 = NBOX * 49;
  for (int h = 0; h < 4; ++h) {
    int nw = HEADN * CIN3P * 9;
    k_pack_w3<<<(nw + 255) / 256, 256, 0, stream>>>((const float*)d_in[h3w1[h]], wPack,
                                                    HEADN, CIN3R, CIN3P);
    k_conv3x3_wmma<1><<<dim3((npix3 + 31) / 32, HEADN / 64, 1), 256, 0, stream>>>(
        roiFull, wPack, (const float*)d_in[h3b1[h]], inter3, CIN3P, 7, 7, npix3, HEADN, 0);
    k_bnpool<<<NBOX, HEADN, 0, stream>>>(inter3, (const float*)d_in[h3g[h]],
                                         (const float*)d_in[h3be[h]],
                                         (const float*)d_in[h3m[h]],
                                         (const float*)d_in[h3v[h]], pooled);
    k_fc<<<NBOX, 32, 0, stream>>>(pooled, (const float*)d_in[h3w2[h]],
                                  (const float*)d_in[h3b2[h]], h3out[h], h3co[h]);
  }

  // ---- depth + scatter s3 ----
  k_final<<<(NBOX + 255) / 256, 256, 0, stream>>>(s3buf, dno, boxesr, calibs, msize,
                                                  out_sz3, out_h3d, out_dep);
}
